// Attention_90168543412906
// MI455X (gfx1250) — compile-verified, matrix operand order checked by reading, not run on hardware
//
#include <hip/hip_runtime.h>
#include <math.h>

#define BATCH 256
#define PIX   196
#define EDIM  2048
#define ADIM  512
#define RS    2056              // LDS row stride in bf16 (2048 + 8 pad -> 4-bank row shift)

typedef __attribute__((ext_vector_type(16))) __bf16 v16bf;
typedef __attribute__((ext_vector_type(8)))  float  v8f;

union Frag { v16bf v; uint4 q[2]; };

__device__ __forceinline__ unsigned short f32_to_bf16(float f) {
  union { float f; unsigned int u; } c; c.f = f;
  unsigned int r = c.u + 0x7FFFu + ((c.u >> 16) & 1u);   // round-to-nearest-even
  return (unsigned short)(r >> 16);
}

// ---------------- W_enc (2048x512 f32) -> Wt (512x2048 bf16), tiled transpose ----------------
__global__ void transpose_wenc(const float* __restrict__ W, unsigned short* __restrict__ Wt) {
  __shared__ float t[16][17];
  int a0 = blockIdx.x * 16;          // over ADIM
  int e0 = blockIdx.y * 16;          // over EDIM
  int tx = threadIdx.x, ty = threadIdx.y;
  t[ty][tx] = W[(size_t)(e0 + ty) * ADIM + (a0 + tx)];
  __syncthreads();
  Wt[(size_t)(a0 + ty) * EDIM + (e0 + tx)] = f32_to_bf16(t[tx][ty]);
}

// ---------------- t2[b][a] = dec[b]·W_dec[:,a] + b_dec[a] + b_enc[a] ----------------
__global__ void t2_kernel(const float* __restrict__ dec, const float* __restrict__ Wd,
                          const float* __restrict__ bdec, const float* __restrict__ benc,
                          float* __restrict__ t2) {
  __shared__ float ds[512];
  int b = blockIdx.x, tid = threadIdx.x;
  ds[tid]       = dec[(size_t)b * 512 + tid];
  ds[tid + 256] = dec[(size_t)b * 512 + tid + 256];
  __syncthreads();
  #pragma unroll
  for (int rep = 0; rep < 2; ++rep) {
    int a = tid + rep * 256;
    float s = 0.f;
    for (int e = 0; e < 512; ++e) s = fmaf(ds[e], Wd[(size_t)e * ADIM + a], s);
    t2[(size_t)b * ADIM + a] = s + bdec[a] + benc[a];
  }
}

// ---------------- fused att1 GEMM (bf16 WMMA) + relu + dot(W_att) -> e[b][p] ----------------
__global__ __launch_bounds__(256)
void attn_main(const float* __restrict__ enc, const unsigned short* __restrict__ Wt,
               const float* __restrict__ t2, const float* __restrict__ Watt,
               const float* __restrict__ batt, float* __restrict__ eout) {
  extern __shared__ unsigned char smem[];
  float* eacc = (float*)smem;                               // 64 floats
  unsigned short* At = (unsigned short*)(smem + 256);       // [64][RS] bf16

  const int b   = blockIdx.y;
  const int bx  = blockIdx.x;              // pixel rows bx*64 .. bx*64+63
  const int tid = threadIdx.x;

  if (tid < 64) eacc[tid] = 0.f;

  // ---- stage A: 64 rows x 2048 f32 -> bf16 in LDS (clamped ghost rows) ----
  for (int it = 0; it < 128; ++it) {                        // 32768 float4 / 256 thr
    int idx = it * 256 + tid;
    int r   = idx >> 9;                                     // 512 float4 per row
    int c4  = idx & 511;
    int p   = bx * 64 + r;
    int pc  = p < PIX ? p : (PIX - 1);
    const float4 v = *(const float4*)(enc + ((size_t)b * PIX + pc) * EDIM + (size_t)c4 * 4);
    ushort4 o;
    o.x = f32_to_bf16(v.x); o.y = f32_to_bf16(v.y);
    o.z = f32_to_bf16(v.z); o.w = f32_to_bf16(v.w);
    *(ushort4*)(At + (size_t)r * RS + (size_t)c4 * 4) = o;
  }
  __syncthreads();

  const int lane  = tid & 31;
  const int wv    = tid >> 5;              // wave 0..7, covers n-tiles wv*4 .. wv*4+3
  const int ll    = lane & 15;
  const int half8 = (lane >> 4) << 3;      // 0 or 8 (ISA 16-bit A/B K split)

  const unsigned short* bp[4];
  #pragma unroll
  for (int nt = 0; nt < 4; ++nt)
    bp[nt] = Wt + (size_t)((wv * 4 + nt) * 16 + ll) * EDIM;

  v8f acc[4][4] = {};                      // [ptile][ntile]

  for (int kb = 0; kb < 64; ++kb) {
    const int K0 = kb * 32;
    Frag bf[4];
    #pragma unroll
    for (int nt = 0; nt < 4; ++nt) {
      bf[nt].q[0] = *(const uint4*)(bp[nt] + K0 + half8);
      bf[nt].q[1] = *(const uint4*)(bp[nt] + K0 + 16 + half8);
    }
    #pragma unroll
    for (int pt = 0; pt < 4; ++pt) {
      const unsigned short* ap = At + (size_t)(pt * 16 + ll) * RS + K0 + half8;
      Frag af;
      af.q[0] = *(const uint4*)ap;         // ds_load_b128
      af.q[1] = *(const uint4*)(ap + 16);
      #pragma unroll
      for (int nt = 0; nt < 4; ++nt)
        acc[pt][nt] = __builtin_amdgcn_wmma_f32_16x16x32_bf16(
            false, af.v, false, bf[nt].v, (short)0, acc[pt][nt], false, false);
    }
  }

  // ---- epilogue: e partial = sum_a relu(att1 + t2) * Watt ----
  float s[4][8] = {};
  #pragma unroll
  for (int nt = 0; nt < 4; ++nt) {
    int a = (wv * 4 + nt) * 16 + ll;
    float tv = t2[(size_t)b * ADIM + a];
    float wa = Watt[a];
    #pragma unroll
    for (int pt = 0; pt < 4; ++pt)
      #pragma unroll
      for (int j = 0; j < 8; ++j) {
        float x = acc[pt][nt][j] + tv;
        x = x > 0.f ? x : 0.f;
        s[pt][j] = fmaf(x, wa, s[pt][j]);
      }
  }
  const int mhi = (lane >> 4) << 3;        // D layout: lanes 0-15 -> M=j, 16-31 -> M=j+8
  #pragma unroll
  for (int pt = 0; pt < 4; ++pt)
    #pragma unroll
    for (int j = 0; j < 8; ++j) {
      float v = s[pt][j];
      v += __shfl_xor(v, 8, 16);
      v += __shfl_xor(v, 4, 16);
      v += __shfl_xor(v, 2, 16);
      v += __shfl_xor(v, 1, 16);
      if (ll == 0) atomicAdd(&eacc[pt * 16 + j + mhi], v);   // ds_add_f32
    }
  __syncthreads();

  if (tid < 64) {
    int p = bx * 64 + tid;
    if (p < PIX) eout[(size_t)b * PIX + p] = eacc[tid] + batt[0];
  }
}

// ---------------- softmax over 196 pixels per batch ----------------
__global__ void softmax_kernel(const float* __restrict__ e, float* __restrict__ alpha) {
  __shared__ float red[256];
  int b = blockIdx.x, tid = threadIdx.x;
  float v = (tid < PIX) ? e[(size_t)b * PIX + tid] : -3.0e38f;
  red[tid] = v; __syncthreads();
  for (int s2 = 128; s2 > 0; s2 >>= 1) {
    if (tid < s2) red[tid] = fmaxf(red[tid], red[tid + s2]);
    __syncthreads();
  }
  float m = red[0]; __syncthreads();
  float ev = (tid < PIX) ? __expf(v - m) : 0.f;
  red[tid] = ev; __syncthreads();
  for (int s2 = 128; s2 > 0; s2 >>= 1) {
    if (tid < s2) red[tid] += red[tid + s2];
    __syncthreads();
  }
  float inv = 1.f / red[0];
  if (tid < PIX) alpha[(size_t)b * PIX + tid] = ev * inv;
}

// ---------------- z[b][e] = sum_p alpha[b][p] * enc[b][p][e] ----------------
__global__ void zsum_kernel(const float* __restrict__ enc, const float* __restrict__ alpha,
                            float* __restrict__ z) {
  __shared__ float al[PIX];
  int b = blockIdx.x, tid = threadIdx.x;
  if (tid < PIX) al[tid] = alpha[(size_t)b * PIX + tid];
  __syncthreads();
  const int e0 = tid * 8;
  float4 z0 = {0, 0, 0, 0}, z1 = {0, 0, 0, 0};
  const float* base = enc + (size_t)b * PIX * EDIM + e0;
  for (int p = 0; p < PIX; ++p) {
    float a = al[p];
    const float4 r0 = *(const float4*)(base + (size_t)p * EDIM);
    const float4 r1 = *(const float4*)(base + (size_t)p * EDIM + 4);
    z0.x = fmaf(a, r0.x, z0.x); z0.y = fmaf(a, r0.y, z0.y);
    z0.z = fmaf(a, r0.z, z0.z); z0.w = fmaf(a, r0.w, z0.w);
    z1.x = fmaf(a, r1.x, z1.x); z1.y = fmaf(a, r1.y, z1.y);
    z1.z = fmaf(a, r1.z, z1.z); z1.w = fmaf(a, r1.w, z1.w);
  }
  *(float4*)(z + (size_t)b * EDIM + e0)     = z0;
  *(float4*)(z + (size_t)b * EDIM + e0 + 4) = z1;
}

extern "C" void kernel_launch(void* const* d_in, const int* in_sizes, int n_in,
                              void* d_out, int out_size, void* d_ws, size_t ws_size,
                              hipStream_t stream) {
  const float* enc  = (const float*)d_in[0];
  const float* dec  = (const float*)d_in[1];
  const float* Wenc = (const float*)d_in[2];
  const float* benc = (const float*)d_in[3];
  const float* Wdec = (const float*)d_in[4];
  const float* bdec = (const float*)d_in[5];
  const float* Watt = (const float*)d_in[6];
  const float* batt = (const float*)d_in[7];

  float* z     = (float*)d_out;                         // 256*2048
  float* alpha = (float*)d_out + (size_t)BATCH * EDIM;  // 256*196

  unsigned short* Wt = (unsigned short*)d_ws;                         // 2 MB bf16
  float* t2   = (float*)((char*)d_ws + (size_t)ADIM * EDIM * 2);      // 256*512
  float* ebuf = t2 + (size_t)BATCH * ADIM;                            // 256*196

  transpose_wenc<<<dim3(ADIM / 16, EDIM / 16), dim3(16, 16), 0, stream>>>(Wenc, Wt);
  t2_kernel<<<BATCH, 256, 0, stream>>>(dec, Wdec, bdec, benc, t2);

  size_t shbytes = 256 + (size_t)64 * RS * 2;           // 263,424 B (< 320 KB/WGP)
  (void)hipFuncSetAttribute((const void*)attn_main,
                            hipFuncAttributeMaxDynamicSharedMemorySize, (int)shbytes);
  attn_main<<<dim3(4, BATCH), 256, shbytes, stream>>>(enc, Wt, t2, Watt, batt, ebuf);

  softmax_kernel<<<BATCH, 256, 0, stream>>>(ebuf, alpha);
  zsum_kernel<<<BATCH, 256, 0, stream>>>(enc, alpha, z);
}
